// SpatialGraphAttention_45853070852145
// MI455X (gfx1250) — matrix operand order repaired and checked by reading, not compiled, not run on hardware
//
#include <hip/hip_runtime.h>
#include <stdint.h>

// ---------------------------------------------------------------------------
// Types for CDNA5 WMMA (wave32, 16x16x32 bf16 -> f32)
// ---------------------------------------------------------------------------
typedef __attribute__((ext_vector_type(16))) __bf16       v16bf;
typedef __attribute__((ext_vector_type(8)))  float        v8f;
typedef __attribute__((ext_vector_type(4)))  unsigned int v4u;
typedef __attribute__((ext_vector_type(4)))  float        v4f;

union FragU { v4u u4[2]; v16bf b; };   // 32 bytes = 16 bf16 = one WMMA A/B operand

__device__ __forceinline__ unsigned short f2bf(float f) {
  unsigned int u = __float_as_uint(f);
  u += 0x7FFFu + ((u >> 16) & 1u);          // round-to-nearest-even
  return (unsigned short)(u >> 16);
}

__device__ __forceinline__ v8f wmma_bf16(const FragU& A, const FragU& B, v8f C) {
  // D = A(16x32) * B(32x16) + C, fp32 accumulate
  return __builtin_amdgcn_wmma_f32_16x16x32_bf16(false, A.b, false, B.b,
                                                 (short)0, C, false, false);
}

// Problem constants
#define BT    32      // B*T
#define NN    512     // nodes
#define DM    512     // d_model
#define NH    8       // heads
#define HD    64      // head dim
#define NQKV  1536

// ---------------------------------------------------------------------------
// fp32 -> bf16 bulk convert (8 elems/thread)
// ---------------------------------------------------------------------------
__global__ __launch_bounds__(256) void k_cvt_bf16(const float* __restrict__ src,
                                                  unsigned short* __restrict__ dst,
                                                  int n8) {
  int i = blockIdx.x * 256 + threadIdx.x;
  if (i >= n8) return;
  v4f a = ((const v4f*)src)[2 * i];
  v4f c = ((const v4f*)src)[2 * i + 1];
  v4u o;
  o.x = (unsigned)f2bf(a.x) | ((unsigned)f2bf(a.y) << 16);
  o.y = (unsigned)f2bf(a.z) | ((unsigned)f2bf(a.w) << 16);
  o.z = (unsigned)f2bf(c.x) | ((unsigned)f2bf(c.y) << 16);
  o.w = (unsigned)f2bf(c.z) | ((unsigned)f2bf(c.w) << 16);
  ((v4u*)dst)[i] = o;
}

// fp32 [K][N] -> bf16 [N][K]  (weights, so each lane's B-operand K-strip is contiguous)
__global__ __launch_bounds__(256) void k_cvt_transpose(const float* __restrict__ src,
                                                       unsigned short* __restrict__ dst,
                                                       int K, int N, int total) {
  int i = blockIdx.x * 256 + threadIdx.x;
  if (i >= total) return;
  int n = i / K, k = i % K;
  dst[i] = f2bf(src[k * N + n]);
}

// ---------------------------------------------------------------------------
// geo bias: g[t][h][m] = geo_embedding[t,m,:] . geo_w[:,h] + geo_b[h]
// ---------------------------------------------------------------------------
__global__ __launch_bounds__(256) void k_geo(const float* __restrict__ geo,
                                             const float* __restrict__ gw,
                                             const float* __restrict__ gb,
                                             float* __restrict__ g) {
  int idx = blockIdx.x * 256 + threadIdx.x;     // 8*8*512 = 32768
  int m = idx & 511, h = (idx >> 9) & 7, t = idx >> 12;
  const float* row = geo + ((size_t)(t * NN + m)) * DM;
  float acc = gb[h];
  for (int k = 0; k < DM; ++k) acc += row[k] * gw[k * NH + h];
  g[idx] = acc;
}

// ---------------------------------------------------------------------------
// dynamic path: conv1d(k=3) + relu + mini-MHA(T=8,D=32,4 heads) + mean + pair_w
// One 256-thread block per (b, n) sequence. p[b][h][n] output (pair_b cancels).
// ---------------------------------------------------------------------------
__global__ __launch_bounds__(256) void k_dyn(const float* __restrict__ dyn,
                                             const float* __restrict__ cw,
                                             const float* __restrict__ cbi,
                                             const float* __restrict__ inw,
                                             const float* __restrict__ inb,
                                             const float* __restrict__ ow,
                                             const float* __restrict__ obi,
                                             const float* __restrict__ pw,
                                             float* __restrict__ p) {
  __shared__ float feats[8][32], hb[8][32], qkv[8][96], lg[4][8][8],
                   obuf[8][32], mo[32], rep[32];
  int b = blockIdx.x >> 9, n = blockIdx.x & 511;
  int tid = threadIdx.x;
  int t = tid >> 5, f = tid & 31;

  feats[t][f] = dyn[(((size_t)(b * 8 + t) * NN) + n) * 32 + f];
  __syncthreads();
  {                                        // conv over T, kernel 3, pad 1
    float acc = cbi[f];
    for (int k = 0; k < 3; ++k) {
      int tt = t + k - 1;
      if (tt >= 0 && tt < 8)
        for (int i = 0; i < 32; ++i) acc += feats[tt][i] * cw[(f * 32 + i) * 3 + k];
    }
    hb[t][f] = fmaxf(acc, 0.f);
  }
  __syncthreads();
  for (int c = tid; c < 768; c += 256) {   // in-proj
    int tt = c / 96, j = c % 96;
    float acc = inb[j];
    for (int i = 0; i < 32; ++i) acc += hb[tt][i] * inw[i * 96 + j];
    qkv[tt][j] = acc;
  }
  __syncthreads();
  {                                        // logits (4 heads, hd=8)
    int hh = tid >> 6, tq = (tid >> 3) & 7, tk = tid & 7;
    float acc = 0.f;
    for (int d = 0; d < 8; ++d) acc += qkv[tq][hh * 8 + d] * qkv[tk][32 + hh * 8 + d];
    lg[hh][tq][tk] = acc * 0.35355339059327373f;   // 1/sqrt(8)
  }
  __syncthreads();
  if (tid < 32) {                          // softmax rows
    int hh = tid >> 3, tq = tid & 7;
    float m = -1e30f;
    for (int k = 0; k < 8; ++k) m = fmaxf(m, lg[hh][tq][k]);
    float s = 0.f;
    for (int k = 0; k < 8; ++k) { float e = __expf(lg[hh][tq][k] - m); lg[hh][tq][k] = e; s += e; }
    float inv = 1.f / s;
    for (int k = 0; k < 8; ++k) lg[hh][tq][k] *= inv;
  }
  __syncthreads();
  {                                        // attn @ v
    int tt = tid >> 5, c = tid & 31, hh = c >> 3, d = c & 7;
    float acc = 0.f;
    for (int k = 0; k < 8; ++k) acc += lg[hh][tt][k] * qkv[k][64 + hh * 8 + d];
    obuf[tt][c] = acc;
  }
  __syncthreads();
  if (tid < 32) {                          // mean over T (commutes with out-proj)
    float acc = 0.f;
    for (int tt = 0; tt < 8; ++tt) acc += obuf[tt][tid];
    mo[tid] = acc * 0.125f;
  }
  __syncthreads();
  if (tid < 32) {                          // out-proj
    float acc = obi[tid];
    for (int i = 0; i < 32; ++i) acc += mo[i] * ow[i * 32 + tid];
    rep[tid] = acc;
  }
  __syncthreads();
  if (tid < 8) {                           // pair projection
    float acc = 0.f;
    for (int d = 0; d < 32; ++d) acc += rep[d] * pw[d * 8 + tid];
    p[((size_t)(b * 8 + tid) * NN) + n] = acc;
  }
}

// Combined per-key softmax bias c[bt][h][m] = g[t][h][m] - p[b][h][m]
// (all query-only terms and pair_b cancel under softmax)
__global__ __launch_bounds__(256) void k_cbias(const float* __restrict__ g,
                                               const float* __restrict__ p,
                                               float* __restrict__ cb) {
  int idx = blockIdx.x * 256 + threadIdx.x;      // 32*8*512 = 131072
  int m = idx & 511, h = (idx >> 9) & 7, bt = idx >> 12;
  int b = bt >> 3, t = bt & 7;
  cb[idx] = g[((t * 8 + h) << 9) + m] - p[((b * 8 + h) << 9) + m];
}

// ---------------------------------------------------------------------------
// QKV GEMM: [16384 x 512](bf16) x wT[1536][512](bf16) + bias.
// 64x64 output block per wave: 4 A-frags + 4 B-frags feed 16 WMMAs per K-step
// (4x arithmetic-intensity of a 16x16 tile -> L2 traffic /4).
// Epilogue scatters to q[bt,h,n,d], k[bt,h,n,d], vT[bt,h,d,n] in bf16.
// ---------------------------------------------------------------------------
__global__ __launch_bounds__(128) void k_gemm_qkv(const unsigned short* __restrict__ xb,
                                                  const unsigned short* __restrict__ wT,
                                                  const float* __restrict__ bias,
                                                  unsigned short* __restrict__ qb,
                                                  unsigned short* __restrict__ kb,
                                                  unsigned short* __restrict__ vt) {
  int wid  = blockIdx.x * 4 + (threadIdx.x >> 5);
  int tm   = (wid / (NQKV / 64)) * 64, tn = (wid % (NQKV / 64)) * 64;
  int lane = threadIdx.x & 31, l15 = lane & 15, hs = lane >> 4;

  const v4u* Ar[4];
  const v4u* Br[4];
  for (int i = 0; i < 4; ++i) {
    Ar[i] = (const v4u*)(xb + (size_t)(tm + i * 16 + l15) * DM);
    Br[i] = (const v4u*)(wT + (size_t)(tn + i * 16 + l15) * DM);
  }
  v8f acc[4][4];
  for (int i = 0; i < 4; ++i)
    for (int j = 0; j < 4; ++j)
      for (int r = 0; r < 8; ++r) acc[i][j][r] = 0.f;

  for (int k0 = 0; k0 < DM / 32; ++k0) {
    FragU A[4], B[4];
    for (int i = 0; i < 4; ++i) {
      A[i].u4[0] = Ar[i][k0 * 4 + hs];     A[i].u4[1] = Ar[i][k0 * 4 + hs + 2];
      B[i].u4[0] = Br[i][k0 * 4 + hs];     B[i].u4[1] = Br[i][k0 * 4 + hs + 2];
    }
    for (int i = 0; i < 4; ++i)
      for (int j = 0; j < 4; ++j)
        acc[i][j] = wmma_bf16(A[i], B[j], acc[i][j]);
  }
  for (int j = 0; j < 4; ++j) {
    int col = tn + j * 16 + l15;
    float bv = bias[col];
    int sec = col >> 9, cc = col & 511, hh = cc >> 6, dd = cc & 63;
    for (int i = 0; i < 4; ++i) {
      for (int r = 0; r < 8; ++r) {
        int row = tm + i * 16 + r + hs * 8;
        int bt = row >> 9, n = row & 511;
        unsigned short val = f2bf(acc[i][j][r] + bv);
        if (sec == 0)      qb[(((size_t)(bt * NH + hh) * NN + n) * HD) + dd] = val;
        else if (sec == 1) kb[(((size_t)(bt * NH + hh) * NN + n) * HD) + dd] = val;
        else               vt[(((size_t)(bt * NH + hh) * HD + dd) * NN) + n] = val;
      }
    }
  }
}

// ---------------------------------------------------------------------------
// Flash attention: 1 wave = 16 queries over all 512 keys, hd=64.
// S = QK^T via 4 WMMAs per 32-key step; online softmax with half-wave shfl
// reductions; P transposed through per-wave LDS slice; O += P*V via 4 WMMAs.
// ---------------------------------------------------------------------------
__global__ __launch_bounds__(256) void k_attn(const unsigned short* __restrict__ qb,
                                              const unsigned short* __restrict__ kb,
                                              const unsigned short* __restrict__ vt,
                                              const float* __restrict__ cb,
                                              unsigned short* __restrict__ ob) {
  __shared__ __align__(16) unsigned short ldsP[8][16][32];  // per-wave P tiles
  int bh = blockIdx.x;                 // bt*8 + h
  int bt = bh >> 3, h = bh & 7;
  int w = threadIdx.x >> 5;
  int lane = threadIdx.x & 31, l15 = lane & 15, hs = lane >> 4;
  int qbase = blockIdx.y * 128 + w * 16;

  // Q fragments (16 rows x 64 d = two K=32 steps)
  FragU Qf0, Qf1;
  {
    const v4u* qr = (const v4u*)(qb + ((size_t)bh * NN + qbase + l15) * HD);
    Qf0.u4[0] = qr[hs];     Qf0.u4[1] = qr[hs + 2];
    Qf1.u4[0] = qr[4 + hs]; Qf1.u4[1] = qr[4 + hs + 2];
  }
  const float* cbr = cb + (size_t)bh * NN;

  float run_max[8], run_sum[8];
  v8f Oacc[4];
  for (int r = 0; r < 8; ++r) { run_max[r] = -1e30f; run_sum[r] = 0.f; }
  for (int dt = 0; dt < 4; ++dt) for (int r = 0; r < 8; ++r) Oacc[dt][r] = 0.f;

  for (int j0 = 0; j0 < NN; j0 += 32) {
    v8f s0, s1;
    for (int r = 0; r < 8; ++r) { s0[r] = 0.f; s1[r] = 0.f; }
    {
      const v4u* kr0 = (const v4u*)(kb + ((size_t)bh * NN + j0 + l15) * HD);
      const v4u* kr1 = (const v4u*)(kb + ((size_t)bh * NN + j0 + 16 + l15) * HD);
      FragU B;
      B.u4[0] = kr0[hs];     B.u4[1] = kr0[hs + 2];     s0 = wmma_bf16(Qf0, B, s0);
      B.u4[0] = kr0[4 + hs]; B.u4[1] = kr0[4 + hs + 2]; s0 = wmma_bf16(Qf1, B, s0);
      B.u4[0] = kr1[hs];     B.u4[1] = kr1[hs + 2];     s1 = wmma_bf16(Qf0, B, s1);
      B.u4[0] = kr1[4 + hs]; B.u4[1] = kr1[4 + hs + 2]; s1 = wmma_bf16(Qf1, B, s1);
    }
    float cb0 = cbr[j0 + l15], cb1 = cbr[j0 + 16 + l15];
    float p0[8], p1[8], alpha[8];
    for (int r = 0; r < 8; ++r) {
      float a = s0[r] * 0.125f + cb0;        // 1/sqrt(64) scale + key bias
      float c = s1[r] * 0.125f + cb1;
      float m = fmaxf(a, c);
      m = fmaxf(m, __shfl_xor(m, 1, 32));
      m = fmaxf(m, __shfl_xor(m, 2, 32));
      m = fmaxf(m, __shfl_xor(m, 4, 32));
      m = fmaxf(m, __shfl_xor(m, 8, 32));
      float mn = fmaxf(run_max[r], m);
      float al = __expf(run_max[r] - mn);
      float e0 = __expf(a - mn), e1 = __expf(c - mn);
      float ts = e0 + e1;
      ts += __shfl_xor(ts, 1, 32);
      ts += __shfl_xor(ts, 2, 32);
      ts += __shfl_xor(ts, 4, 32);
      ts += __shfl_xor(ts, 8, 32);
      run_sum[r] = run_sum[r] * al + ts;
      run_max[r] = mn;
      alpha[r] = al; p0[r] = e0; p1[r] = e1;
    }
    for (int dt = 0; dt < 4; ++dt)
      for (int r = 0; r < 8; ++r) Oacc[dt][r] *= alpha[r];

    // transpose P (C-layout -> A-layout) through this wave's LDS slice
    for (int r = 0; r < 8; ++r) {
      int row = r + hs * 8;
      ldsP[w][row][l15]      = f2bf(p0[r]);
      ldsP[w][row][16 + l15] = f2bf(p1[r]);
    }
    asm volatile("s_wait_dscnt 0" ::: "memory");   // wave-coherent LDS RAW
    FragU Pf;
    {
      const v4u* pr = (const v4u*)&ldsP[w][l15][0];
      Pf.u4[0] = pr[hs * 2]; Pf.u4[1] = pr[hs * 2 + 1];
    }
    for (int dt = 0; dt < 4; ++dt) {
      const v4u* vr = (const v4u*)(vt + ((size_t)bh * HD + dt * 16 + l15) * NN + j0 + hs * 16);
      FragU Vf; Vf.u4[0] = vr[0]; Vf.u4[1] = vr[1];
      Oacc[dt] = wmma_bf16(Pf, Vf, Oacc[dt]);
    }
  }
  // normalize + store bf16 into ob[bt*512+n][h*64+d]
  for (int r = 0; r < 8; ++r) {
    int row = qbase + r + hs * 8;
    float inv = 1.f / run_sum[r];
    for (int dt = 0; dt < 4; ++dt) {
      int d = dt * 16 + l15;
      ob[((size_t)(bt * NN + row)) * DM + h * HD + d] = f2bf(Oacc[dt][r] * inv);
    }
  }
}

// ---------------------------------------------------------------------------
// Output GEMM: ob[16384x512](bf16) x woutT[512][512](bf16) + out_b -> fp32,
// 64x64 block per wave (same blocking as k_gemm_qkv).
// ---------------------------------------------------------------------------
__global__ __launch_bounds__(128) void k_gemm_out(const unsigned short* __restrict__ ab,
                                                  const unsigned short* __restrict__ wT,
                                                  const float* __restrict__ bias,
                                                  float* __restrict__ out) {
  int wid  = blockIdx.x * 4 + (threadIdx.x >> 5);
  int tm   = (wid / (DM / 64)) * 64, tn = (wid % (DM / 64)) * 64;
  int lane = threadIdx.x & 31, l15 = lane & 15, hs = lane >> 4;

  const v4u* Ar[4];
  const v4u* Br[4];
  for (int i = 0; i < 4; ++i) {
    Ar[i] = (const v4u*)(ab + (size_t)(tm + i * 16 + l15) * DM);
    Br[i] = (const v4u*)(wT + (size_t)(tn + i * 16 + l15) * DM);
  }
  v8f acc[4][4];
  for (int i = 0; i < 4; ++i)
    for (int j = 0; j < 4; ++j)
      for (int r = 0; r < 8; ++r) acc[i][j][r] = 0.f;

  for (int k0 = 0; k0 < DM / 32; ++k0) {
    FragU A[4], B[4];
    for (int i = 0; i < 4; ++i) {
      A[i].u4[0] = Ar[i][k0 * 4 + hs];     A[i].u4[1] = Ar[i][k0 * 4 + hs + 2];
      B[i].u4[0] = Br[i][k0 * 4 + hs];     B[i].u4[1] = Br[i][k0 * 4 + hs + 2];
    }
    for (int i = 0; i < 4; ++i)
      for (int j = 0; j < 4; ++j)
        acc[i][j] = wmma_bf16(A[i], B[j], acc[i][j]);
  }
  for (int j = 0; j < 4; ++j) {
    int col = tn + j * 16 + l15;
    float bv = bias[col];
    for (int i = 0; i < 4; ++i)
      for (int r = 0; r < 8; ++r) {
        int row = tm + i * 16 + r + hs * 8;
        out[(size_t)row * DM + col] = acc[i][j][r] + bv;
      }
  }
}

// ---------------------------------------------------------------------------
extern "C" void kernel_launch(void* const* d_in, const int* in_sizes, int n_in,
                              void* d_out, int out_size, void* d_ws, size_t ws_size,
                              hipStream_t stream) {
  (void)in_sizes; (void)n_in; (void)out_size; (void)ws_size;
  const float* x      = (const float*)d_in[0];
  const float* geo    = (const float*)d_in[1];
  const float* dynf   = (const float*)d_in[2];
  const float* qkv_w  = (const float*)d_in[3];
  const float* qkv_b  = (const float*)d_in[4];
  const float* out_w  = (const float*)d_in[5];
  const float* out_b  = (const float*)d_in[6];
  const float* geo_w  = (const float*)d_in[7];
  const float* geo_b  = (const float*)d_in[8];
  const float* conv_w = (const float*)d_in[9];
  const float* conv_b = (const float*)d_in[10];
  const float* in_w   = (const float*)d_in[11];
  const float* in_b   = (const float*)d_in[12];
  const float* mow    = (const float*)d_in[13];
  const float* mob    = (const float*)d_in[14];
  const float* pw     = (const float*)d_in[15];
  // d_in[16] = pair_b: cancels under softmax, unused.

  size_t off = 0;
  char* base = (char*)d_ws;
  auto carve = [&](size_t bytes) -> void* {
    void* q = (void*)(base + off);
    off += (bytes + 255) & ~(size_t)255;
    return q;
  };
  unsigned short* xb    = (unsigned short*)carve((size_t)BT * NN * DM * 2);
  unsigned short* wqkvT = (unsigned short*)carve((size_t)NQKV * DM * 2);
  unsigned short* woutT = (unsigned short*)carve((size_t)DM * DM * 2);
  unsigned short* qb    = (unsigned short*)carve((size_t)BT * NH * NN * HD * 2);
  unsigned short* kb    = (unsigned short*)carve((size_t)BT * NH * NN * HD * 2);
  unsigned short* vt    = (unsigned short*)carve((size_t)BT * NH * HD * NN * 2);
  unsigned short* obuf  = (unsigned short*)carve((size_t)BT * NN * DM * 2);
  float* gbuf = (float*)carve((size_t)8 * NH * NN * 4);
  float* pbuf = (float*)carve((size_t)4 * NH * NN * 4);
  float* cbuf = (float*)carve((size_t)BT * NH * NN * 4);

  // stage bf16 operands
  k_cvt_bf16<<<(BT * NN * DM / 8 + 255) / 256, 256, 0, stream>>>(x, xb, BT * NN * DM / 8);
  k_cvt_transpose<<<(DM * NQKV + 255) / 256, 256, 0, stream>>>(qkv_w, wqkvT, DM, NQKV, DM * NQKV);
  k_cvt_transpose<<<(DM * DM + 255) / 256, 256, 0, stream>>>(out_w, woutT, DM, DM, DM * DM);

  // small bias paths
  k_geo<<<(8 * NH * NN) / 256, 256, 0, stream>>>(geo, geo_w, geo_b, gbuf);
  k_dyn<<<4 * NN, 256, 0, stream>>>(dynf, conv_w, conv_b, in_w, in_b, mow, mob, pw, pbuf);
  k_cbias<<<(BT * NH * NN) / 256, 256, 0, stream>>>(gbuf, pbuf, cbuf);

  // QKV projection (WMMA, 64x64 block/wave): 256 x 24 wave-tiles
  k_gemm_qkv<<<(BT * NN / 64) * (NQKV / 64) / 4, 128, 0, stream>>>(xb, wqkvT, qkv_b, qb, kb, vt);

  // flash attention with folded key-bias (WMMA)
  k_attn<<<dim3(BT * NH, NN / 128), 256, 0, stream>>>(qb, kb, vt, cbuf, obuf);

  // output projection (WMMA, 64x64 block/wave) -> fp32 result
  k_gemm_out<<<(BT * NN / 64) * (DM / 64) / 4, 128, 0, stream>>>(obuf, woutT, out_b, (float*)d_out);
}
